// Retention_53515292508206
// MI455X (gfx1250) — matrix operand order repaired and checked by reading, not compiled, not run on hardware
//
#include <hip/hip_runtime.h>
#include <cstdint>

// ---------------------------------------------------------------------------
// Retention (xPos rotary + decay-masked attention) for MI455X / gfx1250.
// bf16 WMMA (v_wmma_f32_16x16x32_bf16) for all three GEMMs, f32 accumulate.
// TDM (tensor_load_to_lds via inline asm + s_wait_tensorcnt) stages Q/K tiles.
// ---------------------------------------------------------------------------

typedef __bf16 bf16;
typedef __attribute__((ext_vector_type(16))) __bf16 v16bf;
typedef __attribute__((ext_vector_type(8)))  __bf16 v8bf;
typedef __attribute__((ext_vector_type(8)))  float  v8f;

#define BATCH 4
#define SEQ   4096
#define DM    1024
#define HQK   256
#define HV    512

#define LOG2G    (-0.04580368961312479f)   // log2(0.96875)
#define LOG2_1E4 (13.287712379549449f)     // log2(10000)
#define QK_SCALE 0.0625f                   // 256^-0.5

// Device pass for gfx1250 -> use TDM through inline asm (bypasses the clang
// builtin, which __has_builtin rejected on this toolchain).
#if defined(__AMDGCN__) && defined(__gfx1250__)
#define TDM_ASM 1
#else
#define TDM_ASM 0
#endif

#if TDM_ASM
typedef __attribute__((ext_vector_type(4))) unsigned tdm_v4u;
typedef __attribute__((ext_vector_type(8))) unsigned tdm_v8u;

// 1-D contiguous global->LDS DMA of nelem 2-byte elements (bf16).
// D# per CDNA5 ISA ch.8:
//  group0 (4 SGPRs): [1:0]=count=1 | [63:32]=lds_addr | [120:64]=global_addr
//                    | [127:126]=type=2
//  group1 (8 SGPRs): [17:16]=data_size(2B) | [79:48]=tensor_dim0
//                    | [111:80]=tensor_dim1 | [127:112]=tile_dim0
//                    | [207:160]=tensor_dim0_stride
//  groups 2/3 (4 SGPRs each): zero (dims 2..4 unused)
// Must be issued wave-uniform; EXEC is ignored; tracked by TENSORcnt.
static __device__ inline void tdm_load_1d(void* lds_dst, const void* gsrc,
                                          unsigned nelem) {
  unsigned lds_off = (unsigned)(size_t)lds_dst;       // low 32b = LDS offset
  unsigned long long ga = (unsigned long long)(size_t)gsrc;
  tdm_v4u g0;
  tdm_v8u g1;
  tdm_v4u g2 = {0, 0, 0, 0};
  tdm_v4u g3 = {0, 0, 0, 0};
  g0[0] = 1u;                                          // count=1, user mode
  g0[1] = lds_off;                                     // lds_addr
  g0[2] = (unsigned)ga;                                // global_addr[31:0]
  g0[3] = (unsigned)((ga >> 32) & 0x01FFFFFFull)       // global_addr[56:32]
          | (2u << 30);                                // type = 2 ("image")
  g1[0] = (1u << 16);                                  // data_size=1 -> 2B/elem
  g1[1] = (nelem & 0xFFFFu) << 16;                     // tensor_dim0[15:0]
  g1[2] = ((nelem >> 16) & 0xFFFFu)                    // tensor_dim0[31:16]
          | (1u << 16);                                // tensor_dim1 = 1
  g1[3] = (nelem & 0xFFFFu) << 16;                     // tile_dim0
  g1[4] = 0u;                                          // tile_dim1/2 = unused
  g1[5] = nelem;                                       // tensor_dim0_stride lo
  g1[6] = 0u;
  g1[7] = 0u;
  asm volatile("tensor_load_to_lds %0, %1, %2, %3"
               :: "s"(g0), "s"(g1), "s"(g2), "s"(g3)
               : "memory");
}

static __device__ inline void tdm_wait() {
  asm volatile("s_wait_tensorcnt 0x0" ::: "memory");
}
#endif  // TDM_ASM

// A-fragment (16x32 bf16, M x K): lane%16 = row; lane/16 selects K-chunks
// {0..7,16..23} vs {8..15,24..31}.  base -> (row0, k0), row stride ldk elems.
static __device__ inline v16bf load_a_frag(const bf16* base, int ldk, int lane) {
  int row = lane & 15;
  int ko  = (lane >> 4) * 8;
  const bf16* p = base + row * ldk + ko;
  v8bf lo = *(const v8bf*)(p);
  v8bf hi = *(const v8bf*)(p + 16);
  return __builtin_shufflevector(lo, hi, 0,1,2,3,4,5,6,7,8,9,10,11,12,13,14,15);
}

// B-fragment (32x16 bf16, K x N) from TRANSPOSED storage [n][k]:
// lane%16 = column n; lane/16 selects K-chunks {0..7,16..23} vs {8..15,24..31}.
static __device__ inline v16bf load_b_frag(const bf16* baseT, int ldk, int lane) {
  int n  = lane & 15;
  int ko = (lane >> 4) * 8;
  const bf16* p = baseT + n * ldk + ko;
  v8bf lo = *(const v8bf*)(p);
  v8bf hi = *(const v8bf*)(p + 16);
  return __builtin_shufflevector(lo, hi, 0,1,2,3,4,5,6,7,8,9,10,11,12,13,14,15);
}

// ---------------------------------------------------------------------------
// Kernel 0: pack [W_Q | W_K | W_V] -> bf16 Wb[1024][1024]
// ---------------------------------------------------------------------------
__global__ __launch_bounds__(256) void wconvert_kernel(
    const float* __restrict__ Wq, const float* __restrict__ Wk,
    const float* __restrict__ Wv, bf16* __restrict__ Wb) {
  int idx = blockIdx.x * 256 + threadIdx.x;          // 0 .. 1024*1024-1
  int r = idx >> 10, c = idx & 1023;
  float v;
  if (c < 256)      v = Wq[r * 256 + c];
  else if (c < 512) v = Wk[r * 256 + (c - 256)];
  else              v = Wv[r * 512 + (c - 512)];
  Wb[idx] = (bf16)v;
}

// ---------------------------------------------------------------------------
// Kernel 1: projection GEMM (x @ W) with fused xPos rotation.
// 64 rows x 128 cols per workgroup, K-step 32, 8 waves, bf16 WMMA.
// ---------------------------------------------------------------------------
__global__ __launch_bounds__(256) void proj_xpos_kernel(
    const float* __restrict__ x, const bf16* __restrict__ Wb,
    bf16* __restrict__ Qb, bf16* __restrict__ Kb, bf16* __restrict__ Vb) {
  __shared__ alignas(32) bf16 sX[64][32];    // [m][k]
  __shared__ alignas(32) bf16 sW[128][32];   // [n][k] (transposed)

  const int tid  = threadIdx.x;
  const int lane = tid & 31;
  const int wv   = tid >> 5;
  const int rowbase = blockIdx.y * 64;       // over B*S
  const int colbase = blockIdx.x * 128;      // over 1024 QKV cols
  const int mt  = wv >> 1;                   // 0..3
  const int ntb = (wv & 1) * 4;              // 0 or 4

  v8f acc[4];
  for (int j = 0; j < 4; ++j)
    for (int e = 0; e < 8; ++e) acc[j][e] = 0.0f;

  for (int k0 = 0; k0 < DM; k0 += 32) {
    for (int e = tid; e < 64 * 32; e += 256) {
      int r = e >> 5, c = e & 31;
      sX[r][c] = (bf16)x[(size_t)(rowbase + r) * DM + (k0 + c)];
    }
    for (int e = tid; e < 128 * 32; e += 256) {
      int n = e >> 5, k = e & 31;
      sW[n][k] = Wb[(size_t)(k0 + k) * 1024 + (colbase + n)];
    }
    __syncthreads();
    v16bf a = load_a_frag(&sX[mt * 16][0], 32, lane);
    for (int j = 0; j < 4; ++j) {
      v16bf b = load_b_frag(&sW[(ntb + j) * 16][0], 32, lane);
      acc[j] = __builtin_amdgcn_wmma_f32_16x16x32_bf16(
          false, a, false, b, (short)0, acc[j], false, false);
    }
    __syncthreads();
  }

  // xPos rotation: pair (even,odd) cols live in adjacent lanes of C-fragment.
  for (int j = 0; j < 4; ++j) {
    int col = colbase + (ntb + j) * 16 + (lane & 15);
    // per-lane column invariants, hoisted out of the row loop
    int   hd   = col & 255;
    float freq = exp2f(-LOG2_1E4 * (float)(hd & ~1) * (1.0f / 256.0f));
    bool  isQ  = (col < 256);
    bool  even = ((hd & 1) == 0);
    float t    = (even == isQ) ? -1.0f : 1.0f;   // sign of the sin term
    for (int i = 0; i < 8; ++i) {
      int M     = i + ((lane >> 4) << 3);
      int sflat = rowbase + mt * 16 + M;
      int pos   = sflat & (SEQ - 1);
      float val     = acc[j][i];
      float partner = __shfl_xor(val, 1, 32);    // other half of the pair
      if (col < 512) {                           // Q or K head
        float ang = (float)pos * freq;
        float sn = sinf(ang), cs = cosf(ang);
        float o = val * cs + t * partner * sn;
        if (isQ) Qb[(size_t)sflat * HQK + hd] = (bf16)o;
        else     Kb[(size_t)sflat * HQK + hd] = (bf16)o;
      } else {
        Vb[(size_t)sflat * HV + (col - 512)] = (bf16)val;
      }
    }
  }
}

// ---------------------------------------------------------------------------
// helper: stage a 64x256 bf16 K block into LDS (TDM on gfx1250)
// ---------------------------------------------------------------------------
static __device__ inline void stage_k_block(bf16 (*sK)[256], const bf16* src,
                                            int tid, int wv) {
#if TDM_ASM
  (void)tid;
  if (wv == 0) {
    tdm_load_1d(&sK[0][0], src, 64 * 256);
    tdm_wait();
  }
#else
  (void)wv;
  for (int e = tid; e < 64 * 256 / 8; e += 256)
    ((v8bf*)sK)[e] = ((const v8bf*)src)[e];
#endif
}

// ---------------------------------------------------------------------------
// Kernel 2: retention attention, 32 query rows per workgroup, causal key
// blocks of 64.  Pass A: abs-sum of decayed scores.  Pass B: normalized,
// clipped scores @ V.  Decay mask computed in closed form per element.
// ---------------------------------------------------------------------------
__global__ __launch_bounds__(256) void retention_attn_kernel(
    const bf16* __restrict__ Qb, const bf16* __restrict__ Kb,
    const bf16* __restrict__ Vb, float* __restrict__ out) {
  __shared__ alignas(32) bf16 sQ[32][256];    // 16 KB
  __shared__ alignas(32) bf16 sK[64][256];    // 32 KB
  __shared__ alignas(32) bf16 sVt[512][64];   // 64 KB, transposed [v][key]
  __shared__ alignas(32) bf16 sP[32][64];     // 4 KB, score tile (bf16)
  __shared__ float absr[32];
  __shared__ float rn[32];
  __shared__ float inv[32];

  const int tid = threadIdx.x, lane = tid & 31, wv = tid >> 5;
  const int b  = blockIdx.x >> 7;             // 128 query tiles per batch
  const int qt = blockIdx.x & 127;
  const int q0 = qt * 32;

  const bf16* Qbase = Qb + ((size_t)b * SEQ + q0) * HQK;
  const bf16* Kbase = Kb + (size_t)b * SEQ * HQK;
  const bf16* Vbase = Vb + (size_t)b * SEQ * HV;

#if TDM_ASM
  if (wv == 0) {
    tdm_load_1d(&sQ[0][0], Qbase, 32 * 256);
    tdm_wait();
  }
#else
  for (int e = tid; e < 32 * 256 / 8; e += 256)
    ((v8bf*)sQ)[e] = ((const v8bf*)Qbase)[e];
#endif
  if (tid < 32) {
    int pos = q0 + tid;
    // row_sum = (1 - gamma^(pos+1)) / (1 - gamma),  1/(1-gamma) = 32
    float rowsum = 32.0f * (1.0f - exp2f((float)(pos + 1) * LOG2G));
    rn[tid]   = rsqrtf(fmaxf(rowsum, 1.0f));
    absr[tid] = 0.0f;
  }
  __syncthreads();

  const int nblk = (q0 + 95) >> 6;            // causal: keys <= q0+31
  const int mt = wv & 1, nt = wv >> 1;        // score-tile ownership (32x64)

  // ---------------- pass A: abs_sum accumulation ----------------
  for (int blk = 0; blk < nblk; ++blk) {
    int j0 = blk * 64;
    stage_k_block(sK, Kbase + (size_t)j0 * HQK, tid, wv);
    if (blk + 1 < nblk)  // warm L2/WGP$ for the next key block
      __builtin_prefetch(Kbase + (size_t)(j0 + 64) * HQK + tid * 64, 0, 1);
    __syncthreads();
    v8f sc; for (int e = 0; e < 8; ++e) sc[e] = 0.0f;
    for (int kk = 0; kk < 256; kk += 32) {
      v16bf a  = load_a_frag(&sQ[mt * 16][kk], 256, lane);
      v16bf bb = load_b_frag(&sK[nt * 16][kk], 256, lane);
      sc = __builtin_amdgcn_wmma_f32_16x16x32_bf16(
          false, a, false, bb, (short)0, sc, false, false);
    }
    int t_glob = j0 + nt * 16 + (lane & 15);
    for (int i = 0; i < 8; ++i) {
      int m_local = mt * 16 + i + ((lane >> 4) << 3);
      int dist = (q0 + m_local) - t_glob;
      float w = (dist >= 0) ? exp2f((float)dist * LOG2G) : 0.0f;
      float a1 = fabsf(sc[i] * QK_SCALE * w * rn[m_local]);
      a1 += __shfl_xor(a1, 1, 32);
      a1 += __shfl_xor(a1, 2, 32);
      a1 += __shfl_xor(a1, 4, 32);
      a1 += __shfl_xor(a1, 8, 32);            // sum over the 16 N-lanes
      if ((lane & 15) == 0) atomicAdd(&absr[m_local], a1);
    }
    __syncthreads();
  }
  if (tid < 32) inv[tid] = 1.0f / fmaxf(absr[tid], 1.0f);
  __syncthreads();

  // ---------------- pass B: normalized scores @ V ----------------
  const int mtO = wv & 1;                     // output-tile ownership (32x512)
  const int ntO = (wv >> 1) * 8;
  v8f oacc[8];
  for (int j = 0; j < 8; ++j)
    for (int e = 0; e < 8; ++e) oacc[j][e] = 0.0f;

  for (int blk = 0; blk < nblk; ++blk) {
    int j0 = blk * 64;
    stage_k_block(sK, Kbase + (size_t)j0 * HQK, tid, wv);
    // V block, transposed into [v][key] so B-fragments are contiguous
    for (int e = tid; e < 64 * 512 / 8; e += 256) {
      int key = (e * 8) >> 9;
      int c0  = (e * 8) & 511;
      v8bf v = *(const v8bf*)(Vbase + (size_t)(j0 + key) * HV + c0);
      for (int u = 0; u < 8; ++u) sVt[c0 + u][key] = v[u];
    }
    if (blk + 1 < nblk) {
      __builtin_prefetch(Kbase + (size_t)(j0 + 64) * HQK + tid * 64, 0, 1);
      __builtin_prefetch(Vbase + (size_t)(j0 + 64) * HV  + tid * 128, 0, 1);
    }
    __syncthreads();

    v8f sc; for (int e = 0; e < 8; ++e) sc[e] = 0.0f;
    for (int kk = 0; kk < 256; kk += 32) {
      v16bf a  = load_a_frag(&sQ[mt * 16][kk], 256, lane);
      v16bf bb = load_b_frag(&sK[nt * 16][kk], 256, lane);
      sc = __builtin_amdgcn_wmma_f32_16x16x32_bf16(
          false, a, false, bb, (short)0, sc, false, false);
    }
    int t_glob = j0 + nt * 16 + (lane & 15);
    for (int i = 0; i < 8; ++i) {
      int m_local = mt * 16 + i + ((lane >> 4) << 3);
      int dist = (q0 + m_local) - t_glob;
      float w = (dist >= 0) ? exp2f((float)dist * LOG2G) : 0.0f;
      float val = sc[i] * QK_SCALE * w * rn[m_local] * inv[m_local];
      val = fminf(fmaxf(val, -100.0f), 100.0f);
      sP[m_local][nt * 16 + (lane & 15)] = (bf16)val;
    }
    __syncthreads();

    for (int kk = 0; kk < 64; kk += 32) {
      v16bf a = load_a_frag(&sP[mtO * 16][kk], 64, lane);
      for (int j = 0; j < 8; ++j) {
        v16bf bb = load_b_frag(&sVt[(ntO + j) * 16][kk], 64, lane);
        oacc[j] = __builtin_amdgcn_wmma_f32_16x16x32_bf16(
            false, a, false, bb, (short)0, oacc[j], false, false);
      }
    }
    __syncthreads();
  }

  for (int j = 0; j < 8; ++j) {
    int c = (ntO + j) * 16 + (lane & 15);
    for (int i = 0; i < 8; ++i) {
      int s = q0 + mtO * 16 + i + ((lane >> 4) << 3);
      out[((size_t)b * SEQ + s) * HV + c] = oacc[j][i];
    }
  }
}

// ---------------------------------------------------------------------------
extern "C" void kernel_launch(void* const* d_in, const int* in_sizes, int n_in,
                              void* d_out, int out_size, void* d_ws, size_t ws_size,
                              hipStream_t stream) {
  (void)in_sizes; (void)n_in; (void)out_size; (void)ws_size;
  const float* x  = (const float*)d_in[0];
  const float* Wq = (const float*)d_in[1];
  const float* Wk = (const float*)d_in[2];
  const float* Wv = (const float*)d_in[3];
  float* out = (float*)d_out;

  char* ws = (char*)d_ws;
  bf16* Wb = (bf16*)(ws);                               //  2 MB
  bf16* Qb = (bf16*)(ws + (size_t)(2  << 20));          //  8 MB
  bf16* Kb = (bf16*)(ws + (size_t)(10 << 20));          //  8 MB
  bf16* Vb = (bf16*)(ws + (size_t)(18 << 20));          // 16 MB (total 34 MB)

  hipLaunchKernelGGL(wconvert_kernel, dim3(4096), dim3(256), 0, stream,
                     Wq, Wk, Wv, Wb);
  hipLaunchKernelGGL(proj_xpos_kernel, dim3(8, 256), dim3(256), 0, stream,
                     x, Wb, Qb, Kb, Vb);
  hipLaunchKernelGGL(retention_attn_kernel, dim3(BATCH * SEQ / 32), dim3(256),
                     0, stream, Qb, Kb, Vb, out);
}
